// RepSharingKernelNet_4475355922776
// MI455X (gfx1250) — matrix operand order repaired and checked by reading, not compiled, not run on hardware
//
#include <hip/hip_runtime.h>

typedef _Float16 v16h __attribute__((ext_vector_type(16)));
typedef _Float16 v8h  __attribute__((ext_vector_type(8)));
typedef float    v8f  __attribute__((ext_vector_type(8)));

#define HH 1080
#define WW 1920
#define NB 2
#define PXCNT (NB * HH * WW)          // 4,147,200 pixels

// CDNA5 async global->LDS copy (ASYNCcnt path). The toolchain's builtin takes
// pointers to int4 (generic AS), per the compiler diagnostic:
//   'int __attribute__((__vector_size__(4 * sizeof(int)))) *'
#if defined(__AMDGCN__) && __has_builtin(__builtin_amdgcn_global_load_async_to_lds_b128)
#define ASYNC_COPY 1
typedef int v4i_vs __attribute__((__vector_size__(4 * sizeof(int))));
#endif

static __device__ __forceinline__ void wait_async_then_barrier()
{
#if defined(ASYNC_COPY)
#if __has_builtin(__builtin_amdgcn_s_wait_asynccnt)
    __builtin_amdgcn_s_wait_asynccnt(0);
#else
    asm volatile("s_wait_asynccnt 0" ::: "memory");
#endif
#endif
    __syncthreads();
}

// ---------------------------------------------------------------------------
// Kernel 1: pack weights into WMMA A-fragment friendly layout.
// Layout per layer: wts[l][m][K], K = tap*16 + cin, tap 0..8 real, tap 9 zero.
// 16 out-channels (zero padded), 160 K values, f16. Biases zero-padded to 16.
// ---------------------------------------------------------------------------
__global__ void prep_weights(const float* __restrict__ W1, const float* __restrict__ b1,
                             const float* __restrict__ W2, const float* __restrict__ b2,
                             const float* __restrict__ W3, const float* __restrict__ b3,
                             _Float16* __restrict__ wts, float* __restrict__ bias)
{
    int tid = threadIdx.x;
    const float* Ws[3] = {W1, W2, W3};
    const float* bs[3] = {b1, b2, b3};
    const int    Oc[3] = {14, 14, 12};
    const int    Ic[3] = {10, 14, 14};
    for (int t = tid; t < 3 * 16 * 160; t += 256) {
        int l   = t / (16 * 160);
        int r   = t % (16 * 160);
        int m   = r / 160;
        int K   = r % 160;
        int tap = K >> 4;
        int c   = K & 15;
        float v = 0.0f;
        if (m < Oc[l] && c < Ic[l] && tap < 9) {
            int ky = tap / 3, kx = tap % 3;
            v = Ws[l][((m * Ic[l] + c) * 3 + ky) * 3 + kx];
        }
        wts[t] = (_Float16)v;
    }
    for (int t = tid; t < 48; t += 256) {
        int l = t / 16, m = t % 16;
        bias[t] = (m < Oc[l]) ? bs[l][m] : 0.0f;
    }
}

// ---------------------------------------------------------------------------
// Kernel 2: build network input: ch0..2 = gamma(irr*alb), ch3..9 = x_in ch3..9,
// channel-last f16 padded to 16 (32B per pixel -> single b128x2 store).
// ---------------------------------------------------------------------------
__global__ __launch_bounds__(256)
void prep_input(const float* __restrict__ xin, _Float16* __restrict__ out)
{
    long pid = (long)blockIdx.x * 256 + threadIdx.x;   // < PXCNT (divides evenly)
    long b   = pid / ((long)HH * WW);
    long hw  = pid % ((long)HH * WW);
    const float* base = xin + b * 10L * HH * WW + hw;
    float ch[10];
#pragma unroll
    for (int c = 0; c < 10; c++) ch[c] = base[(long)c * HH * WW];

    v16h vo;
#pragma unroll
    for (int c = 0; c < 16; c++) vo[c] = (_Float16)0.0f;
#pragma unroll
    for (int c = 0; c < 3; c++) {
        float v = fmaxf(ch[c] * ch[3 + c], 0.0f);
        vo[c] = (_Float16)__powf(v, 0.45454545454545f);
    }
#pragma unroll
    for (int c = 3; c < 10; c++) vo[c] = (_Float16)ch[c];

    *(v16h*)(out + pid * 16) = vo;
}

// ---------------------------------------------------------------------------
// Kernel 3: 3x3 conv as implicit GEMM via v_wmma_f32_16x16x32_f16.
// Workgroup = 8 waves; output tile = 32 (W) x 16 (H) pixels.
// Each wave: 4 sub-tiles of 16 pixels x 16 out-channels, K = 160 (5 wmma each,
// 20 wmma per wave). Halo tile (34x18 px, ch-last f16x16) staged in LDS via
// CDNA5 async global->LDS b128 copies when available.
// B-frag: lane<16 -> tap 2s, lane>=16 -> tap 2s+1; 16 contiguous f16 from LDS.
// A-frag: 16 contiguous f16 weights, K = 32s + 16*hi + e (same k order as B-frag).
// ---------------------------------------------------------------------------
__global__ __launch_bounds__(256)
void conv3x3_wmma(const _Float16* __restrict__ in, _Float16* __restrict__ out16,
                  float* __restrict__ out12, const _Float16* __restrict__ wts,
                  const float* __restrict__ bias, int storeF32)
{
    __shared__ __attribute__((aligned(32))) _Float16 tile[18 * 34 * 16]; // 19584B

    int tid = threadIdx.x;
    int b   = blockIdx.z;
    int ty0 = blockIdx.y * 16;
    int tx0 = blockIdx.x * 32;

    // cooperative load of 34x18 pixel halo tile, zero padded at image borders
    for (int s = tid; s < 34 * 18; s += 256) {
        int rr = s / 34, cc = s % 34;
        int gy = ty0 - 1 + rr, gx = tx0 - 1 + cc;
        bool inb = (gy >= 0 && gy < HH && gx >= 0 && gx < WW);
        if (inb) {
            long pix = ((long)b * HH + gy) * WW + gx;
            const _Float16* gp = in + pix * 16;
            _Float16* lp = tile + s * 16;
#if defined(ASYNC_COPY)
            __builtin_amdgcn_global_load_async_to_lds_b128(
                (v4i_vs*)(__UINTPTR_TYPE__)gp, (v4i_vs*)(__UINTPTR_TYPE__)lp, 0, 0);
            __builtin_amdgcn_global_load_async_to_lds_b128(
                (v4i_vs*)(__UINTPTR_TYPE__)(gp + 8), (v4i_vs*)(__UINTPTR_TYPE__)(lp + 8), 0, 0);
#else
            *(v16h*)lp = *(const v16h*)gp;
#endif
        } else {
            v16h z;
#pragma unroll
            for (int c = 0; c < 16; c++) z[c] = (_Float16)0.0f;
            *(v16h*)(tile + s * 16) = z;
        }
    }
    wait_async_then_barrier();

    int wave = tid >> 5;
    int lane = tid & 31;
    int col  = lane & 15;
    int hi   = (lane >> 4) & 1;

    // preload 5 A-fragments (weights) in contiguous k order (16*hi + e), matching B
    v16h A[5];
#pragma unroll
    for (int s = 0; s < 5; s++) {
        const _Float16* p = wts + col * 160 + 32 * s + hi * 16;
        v8h lo = *(const v8h*)p;          // K base .. base+7
        v8h hh = *(const v8h*)(p + 8);    // K base+8 .. base+15
        A[s] = __builtin_shufflevector(lo, hh, 0,1,2,3,4,5,6,7,8,9,10,11,12,13,14,15);
    }
    float breg[8];
#pragma unroll
    for (int r = 0; r < 8; r++) breg[r] = bias[r + hi * 8];

#pragma unroll
    for (int q = 0; q < 4; q++) {
        int st = wave * 4 + q;            // sub-tile 0..31
        int cy = st >> 1;                 // row 0..15
        int cx = st & 1;                  // 16-px column half

        v8f acc;
#pragma unroll
        for (int r = 0; r < 8; r++) acc[r] = breg[r];

#pragma unroll
        for (int s = 0; s < 5; s++) {
            int t = 2 * s + hi;
            t = (t > 8) ? 4 : t;          // tap 9 is dummy (zero weights)
            int dy = t / 3 - 1;
            int dx = t - (t / 3) * 3 - 1;
            int idx = ((1 + cy + dy) * 34 + (1 + cx * 16 + col + dx)) * 16;
            v16h Bf = *(const v16h*)(tile + idx);
            acc = __builtin_amdgcn_wmma_f32_16x16x32_f16(
                      false, A[s], false, Bf, (short)0, acc, false, false);
        }

        int gy = ty0 + cy;                // uniform per wave sub-tile
        int gx = tx0 + cx * 16 + col;
        if (gy < HH) {
            long pix = ((long)b * HH + gy) * WW + gx;
            if (!storeF32) {
                v8h h;
#pragma unroll
                for (int r = 0; r < 8; r++) h[r] = (_Float16)fmaxf(acc[r], 0.0f);
                *(v8h*)(out16 + pix * 16 + hi * 8) = h;   // relu, f16, ch-last
            } else {
#pragma unroll
                for (int r = 0; r < 8; r++) {
                    int ch = r + hi * 8;
                    if (ch < 12) out12[pix * 12 + ch] = acc[r];
                }
            }
        }
    }
}

// ---------------------------------------------------------------------------
// Kernel 4: fused guide=exp, softmax alpha, 6 separable box filters in LDS,
// normalized combine, * alb. One 32x32 output tile per workgroup, halo <= 6.
// LDS: 4 raw channels (g, g*irr rgb) + 4 horizontal-sum channels, 44x44 each.
// ---------------------------------------------------------------------------
__global__ __launch_bounds__(256)
void box_combine(const float* __restrict__ outc, const float* __restrict__ xin,
                 float* __restrict__ xout)
{
    __shared__ float sb[8 * 1936];   // 61.9 KB

    int tid = threadIdx.x;
    int b   = blockIdx.z;
    int y0  = blockIdx.y * 32;
    int x0  = blockIdx.x * 32;
    const long HWl = (long)HH * WW;

    float acc[4][3], alpha[4][6], albv[4][3];
#pragma unroll
    for (int j = 0; j < 4; j++) {
        int idx = tid + j * 256;
        int ty = idx >> 5, tx = idx & 31;
        int y = y0 + ty, x = x0 + tx;
        if (y < HH) {
            long pix = ((long)b * HH + y) * WW + x;
            float o[6];
#pragma unroll
            for (int c = 0; c < 6; c++) o[c] = outc[pix * 12 + 6 + c];
            float m = o[0];
#pragma unroll
            for (int c = 1; c < 6; c++) m = fmaxf(m, o[c]);
            float sum = 0.0f;
#pragma unroll
            for (int c = 0; c < 6; c++) { o[c] = __expf(o[c] - m); sum += o[c]; }
            float inv = 1.0f / sum;
#pragma unroll
            for (int c = 0; c < 6; c++) alpha[j][c] = o[c] * inv;
#pragma unroll
            for (int c = 0; c < 3; c++)
                albv[j][c] = xin[((long)b * 10 + 3 + c) * HWl + (long)y * WW + x];
        } else {
#pragma unroll
            for (int c = 0; c < 6; c++) alpha[j][c] = 0.0f;
#pragma unroll
            for (int c = 0; c < 3; c++) albv[j][c] = 0.0f;
        }
#pragma unroll
        for (int c = 0; c < 3; c++) acc[j][c] = 0.0f;
    }

    for (int i = 0; i < 6; i++) {
        int k = 3 + 2 * i;
        int p = i + 1;
        int e = 32 + 2 * p;
        __syncthreads();
        // load region: guide g = exp(out[i]) and g*irr, zero padded
        for (int s = tid; s < e * e; s += 256) {
            int rr = s / e, cc = s % e;
            int gy = y0 - p + rr, gx = x0 - p + cc;
            float g = 0.0f, r0 = 0.0f, r1 = 0.0f, r2 = 0.0f;
            if (gy >= 0 && gy < HH && gx >= 0 && gx < WW) {
                long pix = ((long)b * HH + gy) * WW + gx;
                g = __expf(outc[pix * 12 + i]);
                long off = (long)gy * WW + gx;
                r0 = g * xin[((long)b * 10 + 0) * HWl + off];
                r1 = g * xin[((long)b * 10 + 1) * HWl + off];
                r2 = g * xin[((long)b * 10 + 2) * HWl + off];
            }
            int li = rr * 44 + cc;
            sb[0 * 1936 + li] = g;
            sb[1 * 1936 + li] = r0;
            sb[2 * 1936 + li] = r1;
            sb[3 * 1936 + li] = r2;
        }
        __syncthreads();
        // horizontal box sums (e rows x 32 output cols x 4 channels)
        for (int s = tid; s < e * 32; s += 256) {
            int rr = s >> 5, tx = s & 31;
#pragma unroll
            for (int c = 0; c < 4; c++) {
                const float* row = sb + c * 1936 + rr * 44 + tx;
                float sum = 0.0f;
                for (int dx = 0; dx < k; dx++) sum += row[dx];
                sb[(4 + c) * 1936 + rr * 44 + tx] = sum;
            }
        }
        __syncthreads();
        // vertical box sums + normalized accumulate
#pragma unroll
        for (int j = 0; j < 4; j++) {
            int idx = tid + j * 256;
            int ty = idx >> 5, tx = idx & 31;
            float wsum = 0.0f, n0 = 0.0f, n1 = 0.0f, n2 = 0.0f;
            for (int dy = 0; dy < k; dy++) {
                int li = (ty + dy) * 44 + tx;
                wsum += sb[4 * 1936 + li];
                n0   += sb[5 * 1936 + li];
                n1   += sb[6 * 1936 + li];
                n2   += sb[7 * 1936 + li];
            }
            float inv = 1.0f / wsum;      // invalid pixels have alpha==0
            float a = alpha[j][i];
            acc[j][0] += a * n0 * inv;
            acc[j][1] += a * n1 * inv;
            acc[j][2] += a * n2 * inv;
        }
    }

#pragma unroll
    for (int j = 0; j < 4; j++) {
        int idx = tid + j * 256;
        int ty = idx >> 5, tx = idx & 31;
        int y = y0 + ty, x = x0 + tx;
        if (y < HH) {
#pragma unroll
            for (int c = 0; c < 3; c++)
                xout[(((long)b * 3 + c) * HH + y) * WW + x] = acc[j][c] * albv[j][c];
        }
    }
}

// ---------------------------------------------------------------------------
// Workspace layout:
//   [0, 15360)           packed f16 weights (3 layers x 16 x 160)
//   [15360, 15552)       padded f32 biases (3 x 16)
//   [65536, +132.7MB)    bufX: x_inputs f16x16 ch-last (reused for h2)
//   [next, +132.7MB)     bufH: h1 f16x16 ch-last
//   [next, +199.1MB)     bufO: conv3 output, f32 x 12 ch-last
// ---------------------------------------------------------------------------
extern "C" void kernel_launch(void* const* d_in, const int* in_sizes, int n_in,
                              void* d_out, int out_size, void* d_ws, size_t ws_size,
                              hipStream_t stream)
{
    const float* xin = (const float*)d_in[0];
    const float* W1  = (const float*)d_in[1];
    const float* b1  = (const float*)d_in[2];
    const float* W2  = (const float*)d_in[3];
    const float* b2  = (const float*)d_in[4];
    const float* W3  = (const float*)d_in[5];
    const float* b3  = (const float*)d_in[6];

    char* ws = (char*)d_ws;
    _Float16* wts  = (_Float16*)(ws + 0);
    float*    bias = (float*)(ws + 15360);
    size_t actBytes = (size_t)PXCNT * 32;           // 16 f16 per pixel
    _Float16* bufX = (_Float16*)(ws + 65536);
    _Float16* bufH = (_Float16*)(ws + 65536 + actBytes);
    float*    bufO = (float*)(ws + 65536 + 2 * actBytes);

    prep_weights<<<1, 256, 0, stream>>>(W1, b1, W2, b2, W3, b3, wts, bias);
    prep_input<<<PXCNT / 256, 256, 0, stream>>>(xin, bufX);

    dim3 cg(WW / 32, (HH + 15) / 16, NB);           // 60 x 68 x 2
    conv3x3_wmma<<<cg, 256, 0, stream>>>(bufX, bufH, nullptr, wts + 0,    bias + 0,  0);
    conv3x3_wmma<<<cg, 256, 0, stream>>>(bufH, bufX, nullptr, wts + 2560, bias + 16, 0);
    conv3x3_wmma<<<cg, 256, 0, stream>>>(bufX, nullptr, bufO, wts + 5120, bias + 32, 1);

    dim3 bg(WW / 32, (HH + 31) / 32, NB);           // 60 x 34 x 2
    box_combine<<<bg, 256, 0, stream>>>(bufO, xin, (float*)d_out);
}
